// GAE_3504693313816
// MI455X (gfx1250) — matrix operand order repaired and checked by reading, not compiled, not run on hardware
//
#include <hip/hip_runtime.h>

typedef __bf16 bf16;
typedef __attribute__((ext_vector_type(16))) __bf16 v16bf;
typedef __attribute__((ext_vector_type(8)))  __bf16 v8bf;
typedef __attribute__((ext_vector_type(8)))  float  v8f;

#define NND 8192   // nodes

// ---------------------------------------------------------------- zero ws
__global__ void zero_f4_kernel(float4* __restrict__ p, int n4) {
    int i = blockIdx.x * blockDim.x + threadIdx.x;
    if (i < n4) p[i] = make_float4(0.f, 0.f, 0.f, 0.f);
}

// ---------------------------------------------------------------- degrees
__global__ void degree_kernel(const int* __restrict__ src, const int* __restrict__ dst,
                              float* __restrict__ deg_src, float* __restrict__ deg_dst, int E) {
    int e = blockIdx.x * blockDim.x + threadIdx.x;
    if (e >= E) return;
    __hip_atomic_fetch_add(&deg_src[src[e]], 1.0f, __ATOMIC_RELAXED, __HIP_MEMORY_SCOPE_AGENT);
    __hip_atomic_fetch_add(&deg_dst[dst[e]], 1.0f, __ATOMIC_RELAXED, __HIP_MEMORY_SCOPE_AGENT);
}

// deg -> deg^{-1/2}, zero-degree clamped to 1 (in place)
__global__ void norm_kernel(float* __restrict__ deg, int n) {
    int i = blockIdx.x * blockDim.x + threadIdx.x;
    if (i < n) deg[i] = rsqrtf(fmaxf(deg[i], 1.0f));
}

// ------------------------------------------- W[K][Nc] f32 -> Wt[Nc][K] bf16
__global__ void transpose_bf16_kernel(const float* __restrict__ W, bf16* __restrict__ Wt,
                                      int K, int Nc) {
    int idx = blockIdx.x * blockDim.x + threadIdx.x;
    if (idx >= K * Nc) return;
    int k = idx / Nc, n = idx % Nc;
    Wt[(size_t)n * K + k] = (bf16)W[idx];
}

// ------------------------------- scatter-add: agg[dst] += feat[src]*norm_src[src]
template <int F>
__global__ void scatter_kernel(const float* __restrict__ feat, const int* __restrict__ src,
                               const int* __restrict__ dst, const float* __restrict__ norm_src,
                               float* __restrict__ agg, int E) {
    const int TPE = F / 4;                       // threads per edge, 4 feats each
    int tid = blockIdx.x * blockDim.x + threadIdx.x;
    int e  = tid / TPE;
    int fb = (tid % TPE) * 4;
    if (e >= E) return;
    int s = src[e], d = dst[e];
    float ns = norm_src[s];
    float4 v = *(const float4*)(feat + (size_t)s * F + fb);
    float* out = agg + (size_t)d * F + fb;
    __hip_atomic_fetch_add(out + 0, v.x * ns, __ATOMIC_RELAXED, __HIP_MEMORY_SCOPE_AGENT);
    __hip_atomic_fetch_add(out + 1, v.y * ns, __ATOMIC_RELAXED, __HIP_MEMORY_SCOPE_AGENT);
    __hip_atomic_fetch_add(out + 2, v.z * ns, __ATOMIC_RELAXED, __HIP_MEMORY_SCOPE_AGENT);
    __hip_atomic_fetch_add(out + 3, v.w * ns, __ATOMIC_RELAXED, __HIP_MEMORY_SCOPE_AGENT);
}

// ---------------- WMMA GEMM: out = act( (norm_dst[i]*A[i,:]) @ Wt^T + bias )
// A: [N][K] f32, Wt: [128][K] bf16 (column-major weights), out width = 128.
// One 16x16 tile per wave; 8 waves per block.
__global__ void wmma_gemm_norm(const float* __restrict__ A, const float* __restrict__ norm_dst,
                               const bf16* __restrict__ Wt, const float* __restrict__ bias,
                               float* __restrict__ outF, bf16* __restrict__ outB,
                               int K, int doRelu) {
    const int lane = threadIdx.x & 31;
    const int wave = threadIdx.x >> 5;
    const int tile = blockIdx.x * 8 + wave;      // 512*8 tiles
    const int mt = tile >> 3;
    const int nt = tile & 7;
    const int i0 = mt * 16, j0 = nt * 16;
    const int r   = lane & 15;
    const int hf  = lane >> 4;
    const int kbA = hf * 8;
    const int kbB = hf * 8;
    const float scale = norm_dst[i0 + r];
    const float* arow = A + (size_t)(i0 + r) * K;
    const bf16*  brow = Wt + (size_t)(j0 + r) * K + kbB;
    v8f acc = {};
    for (int ks = 0; ks < K; ks += 32) {
        v16bf a, b;
        const float* ap = arow + ks + kbA;
#pragma unroll
        for (int i = 0; i < 8; ++i) {
            a[i]     = (bf16)(ap[i]      * scale);
            a[8 + i] = (bf16)(ap[16 + i] * scale);
        }
        v8bf blo = *(const v8bf*)(brow + ks);
        v8bf bhi = *(const v8bf*)(brow + ks + 16);
#pragma unroll
        for (int i = 0; i < 8; ++i) { b[i] = blo[i]; b[8 + i] = bhi[i]; }
        acc = __builtin_amdgcn_wmma_f32_16x16x32_bf16(false, a, false, b,
                                                      (short)0, acc, false, false);
    }
    const float bj = bias[j0 + r];
#pragma unroll
    for (int v = 0; v < 8; ++v) {
        float val = acc[v] + bj;
        if (doRelu) val = fmaxf(val, 0.f);
        size_t off = (size_t)(i0 + v + 8 * hf) * 128 + j0 + r;
        if (outF) outF[off] = val;
        else      outB[off] = (bf16)val;
    }
}

// ---------------- decode: out = sigmoid(h @ h^T), h: [8192][128] bf16
// One wave computes a 64(M) x 64(N) register tile: 4 A frags x 4 B frags,
// 16 f32 accumulators (128 VGPRs), 64 WMMAs per wave. All 8 waves in a block
// share the same 64-row M strip, so A-fragment loads hit WGP$/L0.
__global__ void decode_kernel(const bf16* __restrict__ h, float* __restrict__ out) {
    const int lane = threadIdx.x & 31;
    const int wave = threadIdx.x >> 5;
    const int mblk = blockIdx.x >> 4;            // 128 M-blocks of 64 rows
    const int nblk = blockIdx.x & 15;            // 16 N-blocks of 512 cols
    const int i0 = mblk * 64;
    const int j0 = nblk * 512 + wave * 64;
    const int r   = lane & 15;
    const int hf  = lane >> 4;
    const int kbA = hf * 8;
    const int kbB = hf * 8;

    v8f acc[4][4];
#pragma unroll
    for (int mi = 0; mi < 4; ++mi)
#pragma unroll
        for (int nj = 0; nj < 4; ++nj) acc[mi][nj] = v8f{};

#pragma unroll
    for (int ks = 0; ks < 128; ks += 32) {
        v16bf a[4], b[4];
#pragma unroll
        for (int mi = 0; mi < 4; ++mi) {
            const bf16* arow = h + (size_t)(i0 + mi * 16 + r) * 128 + ks + kbA;
            v8bf lo = *(const v8bf*)(arow);
            v8bf hi = *(const v8bf*)(arow + 16);
#pragma unroll
            for (int i = 0; i < 8; ++i) { a[mi][i] = lo[i]; a[mi][8 + i] = hi[i]; }
        }
#pragma unroll
        for (int nj = 0; nj < 4; ++nj) {
            const bf16* brow = h + (size_t)(j0 + nj * 16 + r) * 128 + ks + kbB;
            v8bf lo = *(const v8bf*)(brow);
            v8bf hi = *(const v8bf*)(brow + 16);
#pragma unroll
            for (int i = 0; i < 8; ++i) { b[nj][i] = lo[i]; b[nj][8 + i] = hi[i]; }
        }
#pragma unroll
        for (int mi = 0; mi < 4; ++mi)
#pragma unroll
            for (int nj = 0; nj < 4; ++nj)
                acc[mi][nj] = __builtin_amdgcn_wmma_f32_16x16x32_bf16(
                    false, a[mi], false, b[nj], (short)0, acc[mi][nj], false, false);
    }

#pragma unroll
    for (int mi = 0; mi < 4; ++mi) {
#pragma unroll
        for (int v = 0; v < 8; ++v) {
            const size_t rowoff = (size_t)(i0 + mi * 16 + v + 8 * hf) * 8192;
#pragma unroll
            for (int nj = 0; nj < 4; ++nj) {
                float x  = acc[mi][nj][v];
                float sg = __builtin_amdgcn_rcpf(1.0f + __expf(-x));
                // nontemporal: don't let the 256MB output stream evict L2-resident h
                __builtin_nontemporal_store(sg, out + rowoff + j0 + nj * 16 + r);
            }
        }
    }
}

extern "C" void kernel_launch(void* const* d_in, const int* in_sizes, int n_in,
                              void* d_out, int out_size, void* d_ws, size_t ws_size,
                              hipStream_t stream) {
    const float* x   = (const float*)d_in[0];
    const int*   src = (const int*)d_in[1];
    const int*   dst = (const int*)d_in[2];
    const float* W1  = (const float*)d_in[3];
    const float* b1  = (const float*)d_in[4];
    const float* W2  = (const float*)d_in[5];
    const float* b2  = (const float*)d_in[6];
    float* out = (float*)d_out;
    const int E = in_sizes[1];

    // workspace layout
    float* deg_src = (float*)d_ws;                     // N   (becomes norm_src)
    float* deg_dst = deg_src + NND;                    // N   (becomes norm_dst)
    float* agg1    = deg_dst + NND;                    // N*256
    float* h1      = agg1 + (size_t)NND * 256;         // N*128
    float* agg2    = h1   + (size_t)NND * 128;         // N*128
    bf16*  hb      = (bf16*)(agg2 + (size_t)NND * 128);// N*128 bf16
    bf16*  W1t     = hb  + (size_t)NND * 128;          // 128*256 bf16
    bf16*  W2t     = W1t + 256 * 128;                  // 128*128 bf16

    // zero deg/agg1/h1/agg2 region (contiguous f32 run)
    {
        size_t nfl = (size_t)2 * NND + (size_t)NND * 256 + (size_t)NND * 128 + (size_t)NND * 128;
        int n4 = (int)(nfl / 4);
        zero_f4_kernel<<<(n4 + 255) / 256, 256, 0, stream>>>((float4*)d_ws, n4);
    }
    degree_kernel<<<(E + 255) / 256, 256, 0, stream>>>(src, dst, deg_src, deg_dst, E);
    norm_kernel<<<(2 * NND + 255) / 256, 256, 0, stream>>>(deg_src, 2 * NND);
    transpose_bf16_kernel<<<(256 * 128 + 255) / 256, 256, 0, stream>>>(W1, W1t, 256, 128);
    transpose_bf16_kernel<<<(128 * 128 + 255) / 256, 256, 0, stream>>>(W2, W2t, 128, 128);

    // conv1: scatter-add then fused norm+GEMM+bias+relu
    scatter_kernel<256><<<(E * 64 + 255) / 256, 256, 0, stream>>>(x, src, dst, deg_src, agg1, E);
    wmma_gemm_norm<<<(NND / 16) * 8 / 8, 256, 0, stream>>>(agg1, deg_dst, W1t, b1,
                                                           h1, nullptr, 256, 1);
    // conv2: scatter-add then fused norm+GEMM+bias, emit bf16 h
    scatter_kernel<128><<<(E * 32 + 255) / 256, 256, 0, stream>>>(h1, src, dst, deg_src, agg2, E);
    wmma_gemm_norm<<<(NND / 16) * 8 / 8, 256, 0, stream>>>(agg2, deg_dst, W2t, b2,
                                                           nullptr, hb, 128, 0);
    // decode: sigmoid(h h^T), 8192x8192 f32
    // grid: 128 M-blocks x 16 N-blocks, 8 waves/block, 64x64 tile per wave
    decode_kernel<<<128 * 16, 256, 0, stream>>>(hb, out);
}
